// MultiHeadAttention_44684839747770
// MI455X (gfx1250) — matrix operand order repaired and checked by reading, not compiled, not run on hardware
//
#include <hip/hip_runtime.h>
#include <hip/hip_bf16.h>

// ---------------------------------------------------------------------------
// MultiHeadAttention for MI455X (gfx1250): f16 WMMA pipeline, f32 accumulate.
//   ws:  Qh [N,H,S,D] f16 (pre-scaled by 1/sqrt(D)), Kh [N,H,S,D] f16,
//        Vt [N,H,D,S] f16 (transposed: PV is computed as O^T = V^T * P^T),
//        Ao [N,S,E]   f16 (attention output, input to final projection)
// Flash attention computes S^T = K*Q^T so each lane owns one query column:
// softmax is per-lane + one xor-16 exchange, and P^T feeds the PV WMMA as a
// B-fragment straight from registers (no LDS transpose round-trip at all).
// ---------------------------------------------------------------------------

typedef __attribute__((ext_vector_type(16))) _Float16 v16h;
typedef __attribute__((ext_vector_type(8)))  _Float16 v8h;
typedef __attribute__((ext_vector_type(8)))  float    v8f;
typedef __attribute__((ext_vector_type(4)))  float    v4f;

union V16 { v16h v; v8h h[2]; unsigned int u[8]; };

static __device__ __forceinline__ v8f wmma_f16(const V16& a, const V16& b, v8f c) {
    // D = A(16x32 f16) * B(32x16 f16) + C(16x16 f32)
    return __builtin_amdgcn_wmma_f32_16x16x32_f16(
        /*neg_a=*/false, a.v, /*neg_b=*/false, b.v,
        /*c_mod=*/(short)0, c, /*reuse_a=*/false, /*reuse_b=*/false);
}

static __device__ __forceinline__ unsigned int pk2(float a, float b) {
    union { _Float16 h[2]; unsigned int u; } t;
    t.h[0] = (_Float16)a; t.h[1] = (_Float16)b;
    return t.u;
}

#define NBATCH 2
#define SEQ    2048
#define EMB    1024
#define NHEAD  16
#define HDIM   64

// ---------------------------------------------------------------------------
// Generic projection GEMM: Out = X[Mx1024] @ W^T + bias   (W is [1024,1024])
//   MODE 0: store f16, head-split  [N,H,S,D]      (Q, K)
//   MODE 1: store f16, head-split+T [N,H,D,S]     (V^T)
//   MODE 2: store f32 row-major [M,1024]          (final output)
// Block: 256 threads = 8 waves, tile 128x128, K-step 32.
// ---------------------------------------------------------------------------
template <bool XHALF, int MODE>
__global__ __launch_bounds__(256)
void gemm_proj_kernel(const void* __restrict__ Xv, const float* __restrict__ W,
                      const float* __restrict__ bias, void* __restrict__ Out,
                      float scale)
{
    constexpr int LDA = 40;                  // 32 + 8 halfs pad: conflict-free
    __shared__ _Float16 As[128 * LDA];
    __shared__ _Float16 Bs[128 * LDA];

    const int tid  = threadIdx.x;
    const int lane = tid & 31;
    const int wave = tid >> 5;
    const int wm = wave & 3;                 // 4 waves along M (32 rows each)
    const int wn = wave >> 2;                // 2 waves along N (64 cols each)
    const int bm = blockIdx.x * 128;
    const int bn = blockIdx.y * 128;
    const int hh = lane >> 4;                // lane half
    const int ln = lane & 15;

    v8f acc[2][4];
#pragma unroll
    for (int rt = 0; rt < 2; ++rt)
#pragma unroll
        for (int ct = 0; ct < 4; ++ct)
            acc[rt][ct] = (v8f){0.f,0.f,0.f,0.f,0.f,0.f,0.f,0.f};

    for (int k0 = 0; k0 < EMB; k0 += 32) {
        // ---- stage A tile (128x32) into LDS as f16 ----
        if (XHALF) {
            const _Float16* X = (const _Float16*)Xv;
#pragma unroll
            for (int i = 0; i < 2; ++i) {
                const int idx = tid + i * 256;          // 512 chunks of 8 halfs
                const int row = idx >> 2, c8 = idx & 3;
                v8h d = *(const v8h*)(X + (size_t)(bm + row) * EMB + k0 + c8 * 8);
                *(v8h*)(As + row * LDA + c8 * 8) = d;
            }
        } else {
            const float* X = (const float*)Xv;
#pragma unroll
            for (int i = 0; i < 4; ++i) {
                const int idx = tid + i * 256;          // 1024 float4 chunks
                const int row = idx >> 3, c4 = idx & 7;
                v4f d = *(const v4f*)(X + (size_t)(bm + row) * EMB + k0 + c4 * 4);
                _Float16* p = As + row * LDA + c4 * 4;
                p[0] = (_Float16)d.x; p[1] = (_Float16)d.y;
                p[2] = (_Float16)d.z; p[3] = (_Float16)d.w;
            }
        }
        // ---- stage B tile: rows of W (output features), f32 -> f16 ----
#pragma unroll
        for (int i = 0; i < 4; ++i) {
            const int idx = tid + i * 256;
            const int row = idx >> 3, c4 = idx & 7;
            v4f d = *(const v4f*)(W + (size_t)(bn + row) * EMB + k0 + c4 * 4);
            _Float16* p = Bs + row * LDA + c4 * 4;
            p[0] = (_Float16)d.x; p[1] = (_Float16)d.y;
            p[2] = (_Float16)d.z; p[3] = (_Float16)d.w;
        }
        __syncthreads();

        V16 af[2], bf[4];
#pragma unroll
        for (int rt = 0; rt < 2; ++rt) {
            const _Float16* p = As + (wm * 32 + rt * 16 + ln) * LDA;
            af[rt].h[0] = *(const v8h*)(p + hh * 8);
            af[rt].h[1] = *(const v8h*)(p + 16 + hh * 8);
        }
#pragma unroll
        for (int ct = 0; ct < 4; ++ct) {
            const _Float16* p = Bs + (wn * 64 + ct * 16 + ln) * LDA;
            bf[ct].h[0] = *(const v8h*)(p + hh * 8);
            bf[ct].h[1] = *(const v8h*)(p + 16 + hh * 8);
        }
#pragma unroll
        for (int rt = 0; rt < 2; ++rt)
#pragma unroll
            for (int ct = 0; ct < 4; ++ct)
                acc[rt][ct] = wmma_f16(af[rt], bf[ct], acc[rt][ct]);
        __syncthreads();
    }

    // ---- epilogue: bias, scale, scatter ----
#pragma unroll
    for (int rt = 0; rt < 2; ++rt)
#pragma unroll
        for (int ct = 0; ct < 4; ++ct) {
            const int ecol = bn + wn * 64 + ct * 16 + ln;
            const float bv = bias[ecol];
#pragma unroll
            for (int r = 0; r < 8; ++r) {
                const int mr  = bm + wm * 32 + rt * 16 + r + 8 * hh;
                const float val = (acc[rt][ct][r] + bv) * scale;
                if constexpr (MODE == 2) {
                    ((float*)Out)[(size_t)mr * EMB + ecol] = val;
                } else {
                    const int nb = mr >> 11, s = mr & (SEQ - 1);
                    const int hd = ecol >> 6, dd = ecol & (HDIM - 1);
                    size_t off;
                    if constexpr (MODE == 0)
                        off = (((size_t)(nb * NHEAD + hd)) * SEQ + s) * HDIM + dd;
                    else
                        off = (((size_t)(nb * NHEAD + hd)) * HDIM + dd) * SEQ + s;
                    ((_Float16*)Out)[off] = (_Float16)val;
                }
            }
        }
}

// ---------------------------------------------------------------------------
// Transposed flash attention: grid (S/128, N*H), 256 threads = 8 waves.
// Each wave owns 16 query columns; streams K/V in 64-key chunks.
// S^T = K*Q^T : A-frag = K rows (direct), B-frag = Q rows (direct).
// O^T = V^T*P^T : A-frag = V^T rows (direct), B-frag = P^T from registers.
// No LDS, no __syncthreads; only 10 xor-16 lane exchanges per chunk.
// ---------------------------------------------------------------------------
__global__ __launch_bounds__(256)
void flash_attn_kernel(const _Float16* __restrict__ Qh,   // [N,H,S,D] (pre-scaled)
                       const _Float16* __restrict__ Kh,   // [N,H,S,D]
                       const _Float16* __restrict__ Vt,   // [N,H,D,S]
                       _Float16* __restrict__ Oh)         // [N,S,E] f16
{
    const int tid  = threadIdx.x;
    const int lane = tid & 31;
    const int wave = tid >> 5;
    const int hh = lane >> 4, ln = lane & 15;
    const int nh = blockIdx.y;                            // n*H + h
    const int n  = nh >> 4, h = nh & (NHEAD - 1);
    const int q0 = blockIdx.x * 128 + wave * 16;

    const _Float16* Qb = Qh + (size_t)nh * SEQ * HDIM;
    const _Float16* Kb = Kh + (size_t)nh * SEQ * HDIM;
    const _Float16* Vb = Vt + (size_t)nh * HDIM * SEQ;

    // Q as B-fragments (B^T = Q, row-major): lane = query row q0+ln,
    // chunk cc covers d in [cc*32, cc*32+32), lane half gives d 8hh..+7 and 16+8hh..+7.
    V16 qf[2];
    {
        const _Float16* p = Qb + (size_t)(q0 + ln) * HDIM;
        qf[0].h[0] = *(const v8h*)(p + hh * 8);
        qf[0].h[1] = *(const v8h*)(p + 16 + hh * 8);
        qf[1].h[0] = *(const v8h*)(p + 32 + hh * 8);
        qf[1].h[1] = *(const v8h*)(p + 48 + hh * 8);
    }

    v8f o[4];                                  // O^T tiles: d-tile dt, col = q
#pragma unroll
    for (int t = 0; t < 4; ++t) o[t] = (v8f){0.f,0.f,0.f,0.f,0.f,0.f,0.f,0.f};
    float m = -1e30f, l = 0.f;                 // per-query (per-lane) stats

    for (int kb = 0; kb < SEQ; kb += 64) {
        if (kb + 64 < SEQ) {                   // uniform branch: EXEC untouched
            __builtin_prefetch((const void*)(Kb + (size_t)(kb + 64 + ln) * HDIM), 0, 0);
            __builtin_prefetch((const void*)(Vb + (size_t)ln * SEQ + kb + 64), 0, 0);
        }
        // ---- scores S^T: 4 tiles (16 keys x 16 q), A = K rows (direct) ----
        v8f c[4];
#pragma unroll
        for (int kt = 0; kt < 4; ++kt) {
            const _Float16* kp = Kb + (size_t)(kb + kt * 16 + ln) * HDIM;
            V16 a0, a1;
            a0.h[0] = *(const v8h*)(kp + hh * 8);
            a0.h[1] = *(const v8h*)(kp + 16 + hh * 8);
            a1.h[0] = *(const v8h*)(kp + 32 + hh * 8);
            a1.h[1] = *(const v8h*)(kp + 48 + hh * 8);
            v8f z = (v8f){0.f,0.f,0.f,0.f,0.f,0.f,0.f,0.f};
            z = wmma_f16(a0, qf[0], z);
            c[kt] = wmma_f16(a1, qf[1], z);
        }
        // ---- per-lane online softmax over this lane's 32 keys ----
        float mx = -1e30f;
#pragma unroll
        for (int kt = 0; kt < 4; ++kt)
#pragma unroll
            for (int r = 0; r < 8; ++r) mx = fmaxf(mx, c[kt][r]);
        mx = fmaxf(mx, __shfl_xor(mx, 16));    // other 32 keys: partner half
        const float mn = fmaxf(m, mx);
        const float alpha = __expf(m - mn);
        m = mn;
        float s = 0.f;
#pragma unroll
        for (int kt = 0; kt < 4; ++kt)
#pragma unroll
            for (int r = 0; r < 8; ++r) {
                const float p = __expf(c[kt][r] - mn);
                c[kt][r] = p;
                s += p;
            }
        s += __shfl_xor(s, 16);
        l = l * alpha + s;
#pragma unroll
        for (int t = 0; t < 4; ++t)
#pragma unroll
            for (int r = 0; r < 8; ++r) o[t][r] *= alpha;

        // ---- assemble P^T B-frags: swap 8-key blocks with partner half ----
        V16 pf[2];
#pragma unroll
        for (int cc = 0; cc < 2; ++cc) {
            unsigned int t0[4], t1[4];         // tiles 2cc, 2cc+1 packed f16x2
#pragma unroll
            for (int j = 0; j < 4; ++j) {
                t0[j] = pk2(c[2 * cc][2 * j],     c[2 * cc][2 * j + 1]);
                t1[j] = pk2(c[2 * cc + 1][2 * j], c[2 * cc + 1][2 * j + 1]);
            }
#pragma unroll
            for (int j = 0; j < 4; ++j) {
                const unsigned int sv = hh ? t0[j] : t1[j];
                const unsigned int rx = (unsigned int)__shfl_xor((int)sv, 16);
                pf[cc].u[j]     = hh ? rx    : t0[j];   // keys base..base+7
                pf[cc].u[4 + j] = hh ? t1[j] : rx;      // keys base+8..base+15
            }
        }
        // ---- O^T += V^T * P^T ; A = V^T rows (direct) ----
#pragma unroll
        for (int dt = 0; dt < 4; ++dt) {
            const _Float16* vp = Vb + (size_t)(dt * 16 + ln) * SEQ + kb;
            V16 a0, a1;
            a0.h[0] = *(const v8h*)(vp + hh * 16);
            a0.h[1] = *(const v8h*)(vp + hh * 16 + 8);
            a1.h[0] = *(const v8h*)(vp + 32 + hh * 16);
            a1.h[1] = *(const v8h*)(vp + 32 + hh * 16 + 8);
            o[dt] = wmma_f16(a0, pf[0], o[dt]);
            o[dt] = wmma_f16(a1, pf[1], o[dt]);
        }
    }

    // ---- finalize: lane owns query q0+ln; rows are consecutive d values ----
    const float inv = 1.f / l;
    _Float16* Ob = Oh + ((size_t)n * SEQ + q0 + ln) * EMB + h * HDIM;
#pragma unroll
    for (int dt = 0; dt < 4; ++dt) {
        v8h pkv;
#pragma unroll
        for (int r = 0; r < 8; ++r) pkv[r] = (_Float16)(o[dt][r] * inv);
        *(v8h*)(Ob + dt * 16 + hh * 8) = pkv;   // one 16B store per tile
    }
}

// ---------------------------------------------------------------------------
extern "C" void kernel_launch(void* const* d_in, const int* in_sizes, int n_in,
                              void* d_out, int out_size, void* d_ws, size_t ws_size,
                              hipStream_t stream) {
    const float* key   = (const float*)d_in[0];
    const float* query = (const float*)d_in[1];
    const float* value = (const float*)d_in[2];
    const float* Wk = (const float*)d_in[3];  const float* bk = (const float*)d_in[4];
    const float* Wq = (const float*)d_in[5];  const float* bq = (const float*)d_in[6];
    const float* Wv = (const float*)d_in[7];  const float* bv = (const float*)d_in[8];
    const float* Wo = (const float*)d_in[9];  const float* bo = (const float*)d_in[10];
    float* out = (float*)d_out;

    const size_t TOK = (size_t)NBATCH * SEQ;             // 4096 rows
    _Float16* Qh = (_Float16*)d_ws;
    _Float16* Kh = Qh + TOK * EMB;
    _Float16* Vt = Kh + TOK * EMB;
    _Float16* Ao = Vt + TOK * EMB;

    const dim3 gG(TOK / 128, EMB / 128), bG(256);
    const float qscale = 0.125f;                         // 1/sqrt(64)

    gemm_proj_kernel<false, 0><<<gG, bG, 0, stream>>>(query, Wq, bq, Qh, qscale);
    gemm_proj_kernel<false, 0><<<gG, bG, 0, stream>>>(key,   Wk, bk, Kh, 1.0f);
    gemm_proj_kernel<false, 1><<<gG, bG, 0, stream>>>(value, Wv, bv, Vt, 1.0f);

    flash_attn_kernel<<<dim3(SEQ / 128, NBATCH * NHEAD), 256, 0, stream>>>(Qh, Kh, Vt, Ao);

    gemm_proj_kernel<true, 2><<<gG, bG, 0, stream>>>(Ao, Wo, bo, out, 1.0f);
}
